// RGCN_Block_48198122996158
// MI455X (gfx1250) — matrix operand order repaired and checked by reading, not compiled, not run on hardware
//
#include <hip/hip_runtime.h>

// ---------------------------------------------------------------------------
// RGCN block for MI455X (gfx1250): bf16 WMMA 16x16x32, f32 accumulate.
//   B=32, R=4, N=1024, D=128
//   adjsum[b] = sum_r adj[b,r]          (bf16, computed once)
//   layer: h = x @ W^T + b ; x = h + adjsum @ h
// BK=64 K-tiles through LDS; 8 waves/block, each wave owns a 32x64 C tile.
// ---------------------------------------------------------------------------

typedef __attribute__((ext_vector_type(16))) __bf16 v16bf;
typedef __attribute__((ext_vector_type(8)))  float  v8f;

#define DIMD 128
#define DIMN 1024
#define BK   64
#define LDT  72         // LDS row stride in halfwords (mult of 8 -> 16B aligned runs)

union FragAB { v16bf v; uint4 u[2]; };
union BfBits { __bf16 h; unsigned short s; };

__device__ __forceinline__ unsigned int pack2(float a, float b) {
    // native f32->bf16 converts (RTNE); clang should fuse into v_cvt_pk_bf16_f32
    BfBits lo, hi;
    lo.h = (__bf16)a;
    hi.h = (__bf16)b;
    return (unsigned int)lo.s | ((unsigned int)hi.s << 16);
}

// ---------------------------------------------------------------------------
// Kernel 1: adjsum_bf16[b,n,v] = bf16( sum_r adj[b,r,n,v] )
// ---------------------------------------------------------------------------
__global__ __launch_bounds__(256) void k_adjsum(const float* __restrict__ adj,
                                                unsigned short* __restrict__ adjsum) {
    const size_t NNsz = (size_t)DIMN * DIMN;
    size_t idx = ((size_t)blockIdx.x * 256 + threadIdx.x) * 4;   // element index in [B*N*N)
    size_t b   = idx / NNsz;
    size_t rem = idx - b * NNsz;
    const float* base = adj + b * 4 * NNsz + rem;
    float4 s0 = *(const float4*)(base);
    float4 s1 = *(const float4*)(base + NNsz);
    float4 s2 = *(const float4*)(base + 2 * NNsz);
    float4 s3 = *(const float4*)(base + 3 * NNsz);
    float r0 = s0.x + s1.x + s2.x + s3.x;
    float r1 = s0.y + s1.y + s2.y + s3.y;
    float r2 = s0.z + s1.z + s2.z + s3.z;
    float r3 = s0.w + s1.w + s2.w + s3.w;
    uint2 q;
    q.x = pack2(r0, r1);
    q.y = pack2(r2, r3);
    *(uint2*)(adjsum + idx) = q;
}

// ---------------------------------------------------------------------------
// MMA over one BK=64 slab (two K=32 sub-steps).
// sA: [128][LDT] rows of A (M x K).  sB: [128][LDT] B stored transposed [N][K].
// A frag: lane(l15) = row M, runs at k = ko+half*8 and ko+16+half*8.
// B frag: lane(l15) = col N, runs at k = ko+half*8 and ko+16+half*8.
// ---------------------------------------------------------------------------
__device__ __forceinline__ void mma_step(const unsigned short* sA,
                                         const unsigned short* sB,
                                         int wm, int wn, int half, int l15,
                                         v8f acc[2][4]) {
#pragma unroll
    for (int s = 0; s < 2; ++s) {
        const int ko = s * 32;
        FragAB fa[2], fb[4];
#pragma unroll
        for (int mt = 0; mt < 2; ++mt) {
            int row = wm * 32 + mt * 16 + l15;
            fa[mt].u[0] = *(const uint4*)(&sA[row * LDT + ko + half * 8]);
            fa[mt].u[1] = *(const uint4*)(&sA[row * LDT + ko + 16 + half * 8]);
        }
#pragma unroll
        for (int nt = 0; nt < 4; ++nt) {
            int col = wn * 64 + nt * 16 + l15;
            fb[nt].u[0] = *(const uint4*)(&sB[col * LDT + ko + half * 8]);
            fb[nt].u[1] = *(const uint4*)(&sB[col * LDT + ko + 16 + half * 8]);
        }
#pragma unroll
        for (int mt = 0; mt < 2; ++mt)
#pragma unroll
            for (int nt = 0; nt < 4; ++nt)
                acc[mt][nt] = __builtin_amdgcn_wmma_f32_16x16x32_bf16(
                    false, fa[mt].v, false, fb[nt].v, (short)0, acc[mt][nt],
                    false, false);
    }
}

// ---------------------------------------------------------------------------
// Kernel 2: H[m,e] = sum_d X[m,d]*W[e,d] + bias[e]       M = B*N = 32768
// ---------------------------------------------------------------------------
__global__ __launch_bounds__(256) void k_linear(const float* __restrict__ X,
                                                const float* __restrict__ W,
                                                const float* __restrict__ bias,
                                                float* __restrict__ H) {
    __shared__ __align__(16) unsigned short sA[128 * LDT];
    __shared__ __align__(16) unsigned short sB[128 * LDT];
    __shared__ float sBias[DIMD];

    const int t = threadIdx.x;
    const int mBase = blockIdx.x * 128;
    if (t < DIMD) sBias[t] = bias[t];

    const int lane = t & 31, wave = t >> 5;
    const int wm = wave & 3, wn = wave >> 2;
    const int half = lane >> 4, l15 = lane & 15;
    const int lm = t >> 1;            // 0..127
    const int lk = (t & 1) * 32;      // 0 or 32

    v8f acc[2][4];
    const v8f vzero = {0.f, 0.f, 0.f, 0.f, 0.f, 0.f, 0.f, 0.f};
#pragma unroll
    for (int mt = 0; mt < 2; ++mt)
#pragma unroll
        for (int nt = 0; nt < 4; ++nt) acc[mt][nt] = vzero;

#pragma unroll
    for (int kb = 0; kb < DIMD; kb += BK) {
        __syncthreads();
        // A tile: X rows, convert f32->bf16 (32 elements/thread)
        {
            const float4* ap = (const float4*)(X + (size_t)(mBase + lm) * DIMD + kb + lk);
            float4 f[8];
#pragma unroll
            for (int j = 0; j < 8; ++j) f[j] = ap[j];
#pragma unroll
            for (int j = 0; j < 4; ++j) {
                uint4 q = {pack2(f[2*j].x, f[2*j].y),   pack2(f[2*j].z, f[2*j].w),
                           pack2(f[2*j+1].x, f[2*j+1].y), pack2(f[2*j+1].z, f[2*j+1].w)};
                *(uint4*)(&sA[lm * LDT + lk + 8 * j]) = q;
            }
        }
        // B tile: W[e][d] row-major IS the transposed [N=e][K=d] layout we need
        {
            const float4* wp = (const float4*)(W + (size_t)lm * DIMD + kb + lk);
            float4 f[8];
#pragma unroll
            for (int j = 0; j < 8; ++j) f[j] = wp[j];
#pragma unroll
            for (int j = 0; j < 4; ++j) {
                uint4 q = {pack2(f[2*j].x, f[2*j].y),   pack2(f[2*j].z, f[2*j].w),
                           pack2(f[2*j+1].x, f[2*j+1].y), pack2(f[2*j+1].z, f[2*j+1].w)};
                *(uint4*)(&sB[lm * LDT + lk + 8 * j]) = q;
            }
        }
        __syncthreads();
        mma_step(sA, sB, wm, wn, half, l15, acc);
    }

#pragma unroll
    for (int mt = 0; mt < 2; ++mt)
#pragma unroll
        for (int nt = 0; nt < 4; ++nt) {
            int rowB = mBase + wm * 32 + mt * 16 + half * 8;
            int col  = wn * 64 + nt * 16 + l15;
            float bb = sBias[col];
            float* hp = H + (size_t)rowB * DIMD + col;
#pragma unroll
            for (int i = 0; i < 8; ++i)
                hp[(size_t)i * DIMD] = acc[mt][nt][i] + bb;
        }
}

// ---------------------------------------------------------------------------
// Kernel 3: Out[b,n,e] = Hf[b,n,e] + sum_v adjsum[b,n,v] * Hf[b,v,e]
// grid: x = row block (8), y = batch (32)
// ---------------------------------------------------------------------------
__global__ __launch_bounds__(256) void k_agg(const unsigned short* __restrict__ AS,
                                             const float* __restrict__ Hf,
                                             float* __restrict__ Out) {
    __shared__ __align__(16) unsigned short sA[128 * LDT];
    __shared__ __align__(16) unsigned short sB[128 * LDT];

    const int b = blockIdx.y;
    const int mBase = blockIdx.x * 128;
    const unsigned short* Ab = AS + (size_t)b * DIMN * DIMN;
    const float* Hb = Hf + (size_t)b * DIMN * DIMD;

    const int t = threadIdx.x;
    const int lane = t & 31, wave = t >> 5;
    const int wm = wave & 3, wn = wave >> 2;
    const int half = lane >> 4, l15 = lane & 15;
    const int lm = t >> 1;            // A: row (0..127) | B: column n (0..127)
    const int lk = (t & 1) * 32;      // k sub-offset (0 or 32)

    v8f acc[2][4];
    const v8f vzero = {0.f, 0.f, 0.f, 0.f, 0.f, 0.f, 0.f, 0.f};
#pragma unroll
    for (int mt = 0; mt < 2; ++mt)
#pragma unroll
        for (int nt = 0; nt < 4; ++nt) acc[mt][nt] = vzero;

    for (int kb = 0; kb < DIMN; kb += BK) {
        __syncthreads();
        // A tile: adjsum already bf16; 64B contiguous b128 copies per thread
        {
            const uint4* ap = (const uint4*)(Ab + (size_t)(mBase + lm) * DIMN + kb + lk);
#pragma unroll
            for (int j = 0; j < 4; ++j)
                *(uint4*)(&sA[lm * LDT + lk + 8 * j]) = ap[j];
        }
        // B tile: read h f32 column-slab, convert + store transposed [n][k]
        {
            const float* hp = Hb + (size_t)(kb + lk) * DIMD + lm;
            float f[32];
#pragma unroll
            for (int j = 0; j < 32; ++j) f[j] = hp[(size_t)j * DIMD];
#pragma unroll
            for (int j = 0; j < 4; ++j) {
                uint4 q = {pack2(f[8*j + 0], f[8*j + 1]), pack2(f[8*j + 2], f[8*j + 3]),
                           pack2(f[8*j + 4], f[8*j + 5]), pack2(f[8*j + 6], f[8*j + 7])};
                *(uint4*)(&sB[lm * LDT + lk + 8 * j]) = q;
            }
        }
        __syncthreads();
        mma_step(sA, sB, wm, wn, half, l15, acc);
    }

#pragma unroll
    for (int mt = 0; mt < 2; ++mt)
#pragma unroll
        for (int nt = 0; nt < 4; ++nt) {
            int rowB = mBase + wm * 32 + mt * 16 + half * 8;   // row within batch
            int col  = wn * 64 + nt * 16 + l15;
            const float* hb = Hb + (size_t)rowB * DIMD + col;
            float* op = Out + ((size_t)b * DIMN + rowB) * DIMD + col;
#pragma unroll
            for (int i = 0; i < 8; ++i)
                op[(size_t)i * DIMD] = acc[mt][nt][i] + hb[(size_t)i * DIMD];
        }
}

// ---------------------------------------------------------------------------
extern "C" void kernel_launch(void* const* d_in, const int* in_sizes, int n_in,
                              void* d_out, int out_size, void* d_ws, size_t ws_size,
                              hipStream_t stream) {
    (void)in_sizes; (void)n_in; (void)out_size; (void)ws_size;
    const float* x   = (const float*)d_in[0];
    const float* adj = (const float*)d_in[1];
    const float* W1  = (const float*)d_in[2];
    const float* b1  = (const float*)d_in[3];
    const float* W2  = (const float*)d_in[4];
    const float* b2  = (const float*)d_in[5];
    float* out = (float*)d_out;

    char* ws = (char*)d_ws;
    unsigned short* adjsum = (unsigned short*)ws;                     // 64 MB
    float* h    = (float*)(ws + (size_t)64 * 1024 * 1024);            // 16 MB
    float* xtmp = (float*)(ws + (size_t)80 * 1024 * 1024);            // 16 MB

    // 1) relation reduction, f32 -> bf16
    k_adjsum<<<32768, 256, 0, stream>>>(adj, adjsum);
    // 2) layer 1
    k_linear<<<256, 256, 0, stream>>>(x, W1, b1, h);
    k_agg<<<dim3(8, 32), 256, 0, stream>>>(adjsum, h, xtmp);
    // 3) layer 2
    k_linear<<<256, 256, 0, stream>>>(xtmp, W2, b2, h);
    k_agg<<<dim3(8, 32), 256, 0, stream>>>(adjsum, h, out);
}
